// VisionAttention_29575144800575
// MI455X (gfx1250) — matrix operand order repaired and checked by reading, not compiled, not run on hardware
//
#include <hip/hip_runtime.h>
#include <hip/hip_bf16.h>
#include <math.h>

// ---------------- CDNA5 WMMA types ----------------
typedef _Float16 v16h __attribute__((ext_vector_type(16)));
typedef float    v8f  __attribute__((ext_vector_type(8)));

union Frag {
  v16h  v;
  uint4 u[2];
};

__device__ __forceinline__ v8f vzero8() {
  v8f z;
#pragma unroll
  for (int i = 0; i < 8; ++i) z[i] = 0.0f;
  return z;
}

#define WMMA16(a, b, c) \
  __builtin_amdgcn_wmma_f32_16x16x32_f16(false, (a), false, (b), (short)0, (c), false, false)

// CDNA5 async global->LDS copy (ASYNCcnt-tracked, no VGPR staging).
// Low 32 bits of a generic LDS pointer == LDS byte offset (ISA 10.2).
__device__ __forceinline__ void async_b128(_Float16* lds, const _Float16* g) {
  unsigned s = (unsigned)(unsigned long long)lds;
  asm volatile("global_load_async_to_lds_b128 %0, %1, off"
               :: "v"(s), "v"(g) : "memory");
}
__device__ __forceinline__ void wait_async0() {
  asm volatile("s_wait_asynccnt 0x0" ::: "memory");
}

// ---------------- problem constants ----------------
#define BATCH   8
#define SEQ     1024
#define DMODEL  1152
#define HEADS   16
#define DH      72
#define DHPK    96      // head dim padded for QK^T K-dimension (3 x 32)
#define DHPV    80      // head dim padded for PV output tiles (5 x 16)
#define ROWS    (BATCH * SEQ)   // 8192

// ======================================================================
// Kernel 1: fp32 -> f16 elementwise (hidden states / attention input)
// ======================================================================
__global__ __launch_bounds__(256) void k_cvt_x(const float* __restrict__ X,
                                               _Float16* __restrict__ Xh,
                                               long n) {
  long i = ((long)blockIdx.x * 256 + threadIdx.x) * 4;
  if (i + 3 >= n) return;
#pragma unroll
  for (int u = 0; u < 4; ++u) Xh[i + u] = (_Float16)X[i + u];
}

// ======================================================================
// Kernel 2: transpose + convert weight: Wt[n][k] = (f16) W[k][n]
// ======================================================================
__global__ __launch_bounds__(256) void k_cvt_w(const float* __restrict__ W,
                                               _Float16* __restrict__ Wt) {
  long t = (long)blockIdx.x * 256 + threadIdx.x;   // 0 .. 1152*1152-1
  int n = (int)(t / DMODEL);
  int k = (int)(t % DMODEL);
  Wt[(long)n * DMODEL + k] = (_Float16)W[(long)k * DMODEL + n];
}

// ======================================================================
// Kernel 3: GEMM  C(f32) = A(f16, MxK row-major) * Wt(f16, NxK row-major)^T
// block tile 128x128, 8 waves (4M x 2N) each 32x64, K-step 32.
// Tiles staged via GLOBAL_LOAD_ASYNC_TO_LDS_B128 (ASYNCcnt).
// ======================================================================
__global__ __launch_bounds__(256) void k_gemm(const _Float16* __restrict__ A,
                                              const _Float16* __restrict__ Wt,
                                              float* __restrict__ C,
                                              int M, int N, int K) {
  const int m0 = blockIdx.x * 128;
  const int n0 = blockIdx.y * 128;
  const long z = blockIdx.z;
  const _Float16* Wz = Wt + z * (long)N * K;
  float* Cz = C + z * (long)M * N;

  __shared__ __align__(16) _Float16 Alds[128 * 40];
  __shared__ __align__(16) _Float16 Blds[128 * 40];

  const int tid  = threadIdx.x;
  const int lane = tid & 31;
  const int wave = tid >> 5;
  const int wm   = (wave >> 1) * 32;  // wave M offset in block tile (0..96)
  const int wn   = (wave & 1) * 64;   // wave N offset in block tile (0/64)
  const int lh   = lane >> 4;         // lane half (0/1)
  const int ll   = lane & 15;

  v8f acc[2][4];
#pragma unroll
  for (int i = 0; i < 2; ++i)
#pragma unroll
    for (int j = 0; j < 4; ++j) acc[i][j] = vzero8();

  const int row  = tid >> 1;          // 0..127 (load row for both tiles)
  const int half = tid & 1;           // 32B halves

  const int nk = K / 32;
  for (int kt = 0; kt < nk; ++kt) {
    __syncthreads();  // previous iteration's LDS reads complete
    {  // A tile: 128 rows x 32 cols, async 2 x 16B per thread
      const _Float16* g = A + (long)(m0 + row) * K + kt * 32 + half * 16;
      _Float16* s = Alds + row * 40 + half * 16;
      async_b128(s, g);
      async_b128(s + 8, g + 8);
      if (kt + 1 < nk)  // CDNA5 global_prefetch_b8 on next K tile
        __builtin_prefetch(A + (long)(m0 + row) * K + (kt + 1) * 32, 0, 1);
    }
    {  // B tile (Wt rows): 128 rows x 32 cols, async 2 x 16B per thread
      const _Float16* g = Wz + (long)(n0 + row) * K + kt * 32 + half * 16;
      _Float16* s = Blds + row * 40 + half * 16;
      async_b128(s, g);
      async_b128(s + 8, g + 8);
    }
    wait_async0();
    __syncthreads();

    Frag af[2], bf[4];
#pragma unroll
    for (int ms = 0; ms < 2; ++ms) {  // A frag: row = lane&15, K lo/hi by lane half
      int r = wm + ms * 16 + ll;
      int kb = lh * 8;
      af[ms].u[0] = *(const uint4*)(Alds + r * 40 + kb);
      af[ms].u[1] = *(const uint4*)(Alds + r * 40 + kb + 16);
    }
#pragma unroll
    for (int ns = 0; ns < 4; ++ns) {  // B frag: col = lane&15, K lo/hi by lane half
      int nn = wn + ns * 16 + ll;
      int ko = lh * 8;
      bf[ns].u[0] = *(const uint4*)(Blds + nn * 40 + ko);
      bf[ns].u[1] = *(const uint4*)(Blds + nn * 40 + ko + 16);
    }
#pragma unroll
    for (int ms = 0; ms < 2; ++ms)
#pragma unroll
      for (int ns = 0; ns < 4; ++ns)
        acc[ms][ns] = WMMA16(af[ms].v, bf[ns].v, acc[ms][ns]);
  }

  // C/D layout: VGPR v -> row (v + 8*laneHalf), col = lane&15
#pragma unroll
  for (int ms = 0; ms < 2; ++ms)
#pragma unroll
    for (int ns = 0; ns < 4; ++ns)
#pragma unroll
      for (int v = 0; v < 8; ++v) {
        int r = m0 + wm + ms * 16 + v + 8 * lh;
        int c = n0 + wn + ns * 16 + ll;
        Cz[(long)r * N + c] = acc[ms][ns][v];
      }
}

// ======================================================================
// Kernel 4: per (b,n,h) RMS-norm + 2D RoPE; emit padded f16 Q,K and V^T
// ======================================================================
__global__ __launch_bounds__(256) void k_normrope(
    const float* __restrict__ QKVf, const int* __restrict__ pid,
    const float* __restrict__ qs, const float* __restrict__ ks,
    const float* __restrict__ vs, _Float16* __restrict__ Qr,
    _Float16* __restrict__ Kr, _Float16* __restrict__ Vt) {
  long g = (long)blockIdx.x * 256 + threadIdx.x;  // 0 .. 131071
  if (g >= (long)ROWS * HEADS) return;
  int h   = (int)(g & (HEADS - 1));
  long bn = g >> 4;                 // 0..8191
  int b = (int)(bn >> 10);
  int n = (int)(bn & (SEQ - 1));
  const long bh = (long)b * HEADS + h;

  float p0 = (float)pid[bn * 2 + 0];
  float p1 = (float)pid[bn * 2 + 1];

  // 36 unique (cos,sin): index = k2*18 + freq
  float cb[36], sb[36];
#pragma unroll
  for (int i = 0; i < 36; ++i) {
    int k2 = i / 18, fi = i % 18;
    float invf = __powf(10000.0f, -(float)fi / 18.0f);
    float ang = (k2 ? p1 : p0) * invf;
    cb[i] = __cosf(ang);
    sb[i] = __sinf(ang);
  }

  const float SCAL = 0.11785113019775793f;  // 72^-0.5 folded into Q
  const long MD = (long)ROWS * DMODEL;
  float x[72];

#pragma unroll
  for (int sel = 0; sel < 3; ++sel) {
    const float* src = QKVf + (long)sel * MD + bn * DMODEL + h * DH;
    const float* scl = (sel == 0) ? qs : (sel == 1) ? ks : vs;
    float ss = 0.0f;
#pragma unroll
    for (int j = 0; j < DH; ++j) { x[j] = src[j]; ss += x[j] * x[j]; }
    float r = rsqrtf(ss * (1.0f / DH) + 1e-6f);
#pragma unroll
    for (int j = 0; j < DH; ++j) x[j] *= r * scl[j];

    if (sel == 2) {  // V: no rope, store transposed [B*H][80][1024]
      _Float16* dst = Vt + (bh * DHPV) * SEQ + n;
#pragma unroll
      for (int j = 0; j < DH; ++j) dst[(long)j * SEQ] = (_Float16)x[j];
#pragma unroll
      for (int j = DH; j < DHPV; ++j) dst[(long)j * SEQ] = (_Float16)0.0f;
    } else {  // Q/K: 2D rope, store [B*H][1024][96]
      _Float16* dst = (sel == 0 ? Qr : Kr) + (bh * SEQ + n) * DHPK;
      float mul = (sel == 0) ? SCAL : 1.0f;
#pragma unroll
      for (int j = 0; j < DH; ++j) {
        int t = j % 36;
        int idx = (j / 36) * 18 + (j % 18);
        float rh = (t < 18) ? -x[j + 18] : x[j - 18];
        dst[j] = (_Float16)((x[j] * cb[idx] + rh * sb[idx]) * mul);
      }
#pragma unroll
      for (int j = DH; j < DHPK; ++j) dst[j] = (_Float16)0.0f;
    }
  }
}

// ======================================================================
// Kernel 5: flash attention. grid = (B*H, SEQ/64), 4 waves x 16 q-rows.
// ======================================================================
__global__ __launch_bounds__(128) void k_attn(const _Float16* __restrict__ Qr,
                                              const _Float16* __restrict__ Kr,
                                              const _Float16* __restrict__ Vt,
                                              _Float16* __restrict__ AO) {
  const int bh = blockIdx.x;         // 0..127
  const int b = bh >> 4, h = bh & 15;
  const int qt = blockIdx.y;         // 0..15
  const int tid = threadIdx.x, lane = tid & 31, wave = tid >> 5;
  const int lh = lane >> 4, ll = lane & 15;

  __shared__ __align__(16) _Float16 KtL[32 * DHPK];   // 32 keys x 96
  __shared__ __align__(16) _Float16 VtL[DHPV * 32];   // 80 dh-rows x 32 keys
  __shared__ __align__(16) _Float16 PL[4 * 16 * 32];  // per-wave P strip

  const int q0 = qt * 64 + wave * 16;
  const long qkbase = (long)bh * SEQ;

  Frag aq[3];  // Q A-fragments over K = 96
#pragma unroll
  for (int ksi = 0; ksi < 3; ++ksi) {
    const _Float16* g = Qr + (qkbase + q0 + ll) * DHPK + ksi * 32 + lh * 8;
    aq[ksi].u[0] = *(const uint4*)g;
    aq[ksi].u[1] = *(const uint4*)(g + 16);
  }

  v8f accO[5];
#pragma unroll
  for (int d = 0; d < 5; ++d) accO[d] = vzero8();
  float rmax[8], rsum[8];
#pragma unroll
  for (int v = 0; v < 8; ++v) { rmax[v] = -INFINITY; rsum[v] = 0.0f; }

  for (int kt = 0; kt < SEQ / 32; ++kt) {
    __syncthreads();  // previous iteration's LDS reads complete
    {  // K tile: 32 rows x 96 f16; 4 threads/row, 3 async 16B chunks
      int row = tid >> 2, part = tid & 3;
      const _Float16* g = Kr + (qkbase + kt * 32 + row) * DHPK + part * 24;
      _Float16* s = KtL + row * DHPK + part * 24;
      async_b128(s, g);
      async_b128(s + 8, g + 8);
      async_b128(s + 16, g + 16);
    }
    {  // V^T tile: 80 rows x 32 keys; 160 x 16B async chunks
      for (int i = tid; i < 160; i += 128) {
        int row = i >> 1, half = i & 1;
        async_b128(VtL + row * 32 + half * 8,
                   Vt + ((long)bh * DHPV + row) * SEQ + kt * 32 + half * 8);
      }
    }
    wait_async0();
    __syncthreads();

    // S = Q * K^T : 16 q-rows x 32 keys per wave, K-dim 96
    v8f s0 = vzero8(), s1 = vzero8();
#pragma unroll
    for (int ksi = 0; ksi < 3; ++ksi) {
      int koff = ksi * 32 + lh * 8;
      Frag bk0, bk1;
      bk0.u[0] = *(const uint4*)(KtL + ll * DHPK + koff);
      bk0.u[1] = *(const uint4*)(KtL + ll * DHPK + koff + 16);
      bk1.u[0] = *(const uint4*)(KtL + (16 + ll) * DHPK + koff);
      bk1.u[1] = *(const uint4*)(KtL + (16 + ll) * DHPK + koff + 16);
      s0 = WMMA16(aq[ksi].v, bk0.v, s0);
      s1 = WMMA16(aq[ksi].v, bk1.v, s1);
    }

    // online softmax (row reductions across 16-lane groups)
#pragma unroll
    for (int v = 0; v < 8; ++v) {
      float a = s0[v], c = s1[v];
      float t = fmaxf(a, c);
#pragma unroll
      for (int o = 1; o < 16; o <<= 1) t = fmaxf(t, __shfl_xor(t, o, 32));
      float mn = fmaxf(rmax[v], t);
      float corr = __expf(rmax[v] - mn);
      float pa = __expf(a - mn), pc = __expf(c - mn);
      float ps = pa + pc;
#pragma unroll
      for (int o = 1; o < 16; o <<= 1) ps += __shfl_xor(ps, o, 32);
      rsum[v] = rsum[v] * corr + ps;
      rmax[v] = mn;
#pragma unroll
      for (int d = 0; d < 5; ++d) accO[d][v] *= corr;
      int m = v + 8 * lh;
      PL[wave * 512 + m * 32 + ll]      = (_Float16)pa;
      PL[wave * 512 + m * 32 + 16 + ll] = (_Float16)pc;
    }
    __syncthreads();  // P visible within wave (LDS crosses lanes)

    // O += P * V : A-frag of P, B-frags from V^T tile, K-dim 32
    Frag pf;
    pf.u[0] = *(const uint4*)(PL + wave * 512 + ll * 32 + lh * 8);
    pf.u[1] = *(const uint4*)(PL + wave * 512 + ll * 32 + lh * 8 + 16);
#pragma unroll
    for (int d = 0; d < 5; ++d) {
      Frag vb;
      vb.u[0] = *(const uint4*)(VtL + (d * 16 + ll) * 32 + lh * 8);
      vb.u[1] = *(const uint4*)(VtL + (d * 16 + ll) * 32 + lh * 8 + 16);
      accO[d] = WMMA16(pf.v, vb.v, accO[d]);
    }
  }

  // normalize + write f16 [B][N][H*72] for the output projection
#pragma unroll
  for (int d = 0; d < 5; ++d) {
    int dh = d * 16 + ll;
    if (dh < DH) {
#pragma unroll
      for (int v = 0; v < 8; ++v) {
        int q = q0 + v + 8 * lh;
        AO[((long)(b * SEQ + q)) * DMODEL + h * DH + dh] =
            (_Float16)(accO[d][v] / rsum[v]);
      }
    }
  }
}

// ======================================================================
// launcher
// ======================================================================
extern "C" void kernel_launch(void* const* d_in, const int* in_sizes, int n_in,
                              void* d_out, int out_size, void* d_ws, size_t ws_size,
                              hipStream_t stream) {
  const float* X   = (const float*)d_in[0];
  const int*   pid = (const int*)d_in[1];
  const float* Wq  = (const float*)d_in[2];
  const float* Wk  = (const float*)d_in[3];
  const float* Wv  = (const float*)d_in[4];
  const float* Wo  = (const float*)d_in[5];
  const float* qs  = (const float*)d_in[6];
  const float* ksc = (const float*)d_in[7];
  const float* vsc = (const float*)d_in[8];
  float* OUT = (float*)d_out;

  const size_t WW = (size_t)DMODEL * DMODEL;          // 1152*1152
  const size_t MD = (size_t)ROWS * DMODEL;            // 8192*1152

  char* w = (char*)d_ws;
  _Float16* Xh    = (_Float16*)w;  w += MD * 2;
  _Float16* WtAll = (_Float16*)w;  w += 4 * WW * 2;   // q,k,v,o transposed f16
  float*    QKVf  = (float*)w;     w += 3 * MD * 4;   // fp32 projections
  _Float16* Qr    = (_Float16*)w;  w += (size_t)BATCH * HEADS * SEQ * DHPK * 2;
  _Float16* Kr    = (_Float16*)w;  w += (size_t)BATCH * HEADS * SEQ * DHPK * 2;
  _Float16* Vt    = (_Float16*)w;  w += (size_t)BATCH * HEADS * DHPV * SEQ * 2;
  _Float16* AO    = (_Float16*)w;  w += MD * 2;

  // 1) converts
  k_cvt_x<<<(unsigned)(MD / 4 / 256), 256, 0, stream>>>(X, Xh, (long)MD);
  const unsigned wblocks = (unsigned)(WW / 256);
  k_cvt_w<<<wblocks, 256, 0, stream>>>(Wq, WtAll + 0 * WW);
  k_cvt_w<<<wblocks, 256, 0, stream>>>(Wk, WtAll + 1 * WW);
  k_cvt_w<<<wblocks, 256, 0, stream>>>(Wv, WtAll + 2 * WW);
  k_cvt_w<<<wblocks, 256, 0, stream>>>(Wo, WtAll + 3 * WW);

  // 2) fused QKV projection (z = 0,1,2 selects weight/output)
  k_gemm<<<dim3(ROWS / 128, DMODEL / 128, 3), 256, 0, stream>>>(
      Xh, WtAll, QKVf, ROWS, DMODEL, DMODEL);

  // 3) RMS norm + 2D rope + layout for attention
  k_normrope<<<(ROWS * HEADS) / 256, 256, 0, stream>>>(
      QKVf, pid, qs, ksc, vsc, Qr, Kr, Vt);

  // 4) flash attention
  k_attn<<<dim3(BATCH * HEADS, SEQ / 64), 128, 0, stream>>>(Qr, Kr, Vt, AO);

  // 5) output projection -> d_out (fp32)
  k_gemm<<<dim3(ROWS / 128, DMODEL / 128, 1), 256, 0, stream>>>(
      AO, WtAll + 3 * WW, OUT, ROWS, DMODEL, DMODEL);
}
